// CrossTemporalAttention_16673063043431
// MI455X (gfx1250) — compile-verified, matrix operand order checked by reading, not run on hardware
//
#include <hip/hip_runtime.h>

typedef _Float16 f16;
typedef __attribute__((ext_vector_type(8)))  _Float16 v8h;
typedef __attribute__((ext_vector_type(16))) _Float16 v16h;
typedef __attribute__((ext_vector_type(8)))  float    v8f;
typedef __attribute__((ext_vector_type(4)))  int      v4i;

constexpr int kB = 4;    // batch
constexpr int kC = 256;  // channels / head dim
constexpr int kN = 4096; // tokens = 64*64

#if __has_builtin(__builtin_amdgcn_global_load_async_to_lds_b128)
#define ASYNC_LDS 1
#endif

// Build a 16-half A/B WMMA fragment from two 16-byte chunks:
// per-lane K order is {k0..k0+7, k0+16..k0+23}, caller passes the two bases.
__device__ __forceinline__ v16h make_frag(const f16* p0, const f16* p1) {
  v8h lo = *(const v8h*)p0;
  v8h hi = *(const v8h*)p1;
  return __builtin_shufflevector(lo, hi, 0, 1, 2, 3, 4, 5, 6, 7,
                                 8, 9, 10, 11, 12, 13, 14, 15);
}

// 16-byte global->LDS copy. Async path uses the CDNA5 async pipe
// (GLOBAL_LOAD_ASYNC_TO_LDS_B128, tracked by ASYNCcnt); fallback is a
// synchronous VGPR round trip. The builtin wants int4* in AS(1)/AS(3);
// pointers are built via integer casts (low 32 bits of a generic LDS
// pointer are the LDS byte offset per the aperture mapping).
__device__ __forceinline__ void copy_b128_to_lds(const f16* g, f16* l) {
#ifdef ASYNC_LDS
  __builtin_amdgcn_global_load_async_to_lds_b128(
      (__attribute__((address_space(1))) v4i*)(unsigned long long)(const void*)g,
      (__attribute__((address_space(3))) v4i*)(unsigned int)(unsigned long long)(void*)l,
      0, 0);
#else
  *(v8h*)l = *(const v8h*)g;
#endif
}

__device__ __forceinline__ void wait_async_copies() {
#ifdef ASYNC_LDS
#if __has_builtin(__builtin_amdgcn_s_wait_asynccnt)
  __builtin_amdgcn_s_wait_asynccnt(0);
#else
  asm volatile("s_wait_asynccnt 0" ::: "memory");
#endif
#endif
}

// ---------------------------------------------------------------------------
// Kernel 0: convert the three f32 weight matrices to f16 once.
// ---------------------------------------------------------------------------
__global__ void wcvt_kernel(const float* __restrict__ Wq,
                            const float* __restrict__ Wk,
                            const float* __restrict__ Wv,
                            f16* __restrict__ Wh) {
  int i = blockIdx.x * blockDim.x + threadIdx.x;  // 0 .. 3*65536-1
  const float* src = (i < 65536) ? Wq : (i < 131072 ? Wk : Wv);
  Wh[i] = (f16)src[i & 65535];
}

// ---------------------------------------------------------------------------
// Kernel 1: fused QKV projection for both streams.
//   x [B][C][N] f32  ->  Qh,Kh [s][b][n][c] f16,  Vt [s][b][d][n] f16
// Block: 128 threads (4 waves), 64 tokens per block. Each wave: 16 tokens.
// ---------------------------------------------------------------------------
#define TPITCH 264  // 256 + 8 halfs, keeps rows 16B aligned

__global__ __launch_bounds__(128) void proj_kernel(
    const float* __restrict__ x1, const float* __restrict__ x2,
    const f16* __restrict__ Wh,  // [3][256][256] f16
    const float* __restrict__ bq, const float* __restrict__ bk,
    const float* __restrict__ bv,
    f16* __restrict__ Qh, f16* __restrict__ Kh, f16* __restrict__ Vt) {
  __shared__ f16 tl[64 * TPITCH];

  const int tid = threadIdx.x;
  const int lane = tid & 31, wave = tid >> 5;
  const int half = lane >> 4, l16 = lane & 15;
  const int kb = half * 8;
  const int b = blockIdx.y, s = blockIdx.z;
  const int n0 = blockIdx.x * 64;
  const float* x = (s ? x2 : x1) + (size_t)b * kC * kN;

  // Stage transposed token tile: tl[n][c] = x[c][n0+n], f32 -> f16.
  for (int i = tid; i < kC * 16; i += 128) {
    int c = i >> 4, n4 = (i & 15) * 4;
    float4 v = *(const float4*)(x + (size_t)c * kN + n0 + n4);
    f16* dst = &tl[n4 * TPITCH + c];
    dst[0 * TPITCH] = (f16)v.x;
    dst[1 * TPITCH] = (f16)v.y;
    dst[2 * TPITCH] = (f16)v.z;
    dst[3 * TPITCH] = (f16)v.w;
  }
  __syncthreads();

  const int row0 = wave * 16;
  v16h a[8];
  const f16* ar = &tl[(row0 + l16) * TPITCH];
#pragma unroll
  for (int ks = 0; ks < 8; ++ks) {
    int cb = ks * 32 + kb;
    a[ks] = make_frag(ar + cb, ar + cb + 16);
  }

  const size_t sb = (size_t)(s * kB + b);
  const float* biases[3] = {bq, bk, bv};

  for (int mat = 0; mat < 3; ++mat) {
    const f16* W = Wh + (size_t)mat * kC * kC;
#pragma unroll 2
    for (int dt = 0; dt < 16; ++dt) {
      int d = dt * 16 + l16;
      const f16* wr = W + (size_t)d * kC;  // B[c][d] = W[d][c], contiguous c
      v8f acc = {};
#pragma unroll
      for (int ks = 0; ks < 8; ++ks) {
        int cb = ks * 32 + kb;
        v16h bfr = make_frag(wr + cb, wr + cb + 16);
        acc = __builtin_amdgcn_wmma_f32_16x16x32_f16(
            false, a[ks], false, bfr, (short)0, acc, false, false);
      }
      float bias = biases[mat][d];
      if (mat == 2) {
        // V stored transposed [d][n]: rows r are consecutive n -> one b128.
        v8h pk;
#pragma unroll
        for (int r = 0; r < 8; ++r) pk[r] = (f16)(acc[r] + bias);
        *(v8h*)(Vt + (sb * kC + d) * (size_t)kN + n0 + row0 + half * 8) = pk;
      } else {
        f16* dst = (mat == 0 ? Qh : Kh);
#pragma unroll
        for (int r = 0; r < 8; ++r) {
          int n = n0 + row0 + r + half * 8;
          dst[(sb * kN + n) * (size_t)kC + d] = (f16)(acc[r] + bias);
        }
      }
    }
  }
}

// ---------------------------------------------------------------------------
// Kernel 2: flash attention with async double-buffered K/V tile staging.
//   z=0: softmax(Q1 K2^T) V1 ; z=1: softmax(Q2 K1^T) V2
// Block: 256 threads (8 waves), 128 query rows per block; 32-key tiles.
// ---------------------------------------------------------------------------
#define KP 264  // K-tile row pitch (halfs), 16B aligned rows
#define VP 40   // V-tile row pitch (halfs)
#define PP 40   // P-scratch row pitch (halfs), conflict-free

constexpr int kKtHalfs = 32 * KP;        // 8448
constexpr int kVlHalfs = kC * VP;        // 10240
constexpr int kPlHalfs = 8 * 16 * PP;    // 5120
constexpr int kSmemBytes = (2 * kKtHalfs + 2 * kVlHalfs + kPlHalfs) * 2;

// Issue async fills of one 32-key K tile and matching V^T tile.
__device__ __forceinline__ void stage_tiles(const f16* __restrict__ Kb,
                                            const f16* __restrict__ Vb,
                                            int m0, f16* Kt, f16* Vl, int tid) {
#pragma unroll
  for (int t = 0; t < 4; ++t) {
    int i = tid + t * 256;  // 1024 v8h chunks: 32 keys x 256 c
    int kk = i >> 5, off = (i & 31) * 8;
    copy_b128_to_lds(Kb + (size_t)(m0 + kk) * kC + off, &Kt[kk * KP + off]);
  }
#pragma unroll
  for (int t = 0; t < 4; ++t) {
    int i = tid + t * 256;  // 1024 v8h chunks: 256 d x 32 m
    int d = i >> 2, off = (i & 3) * 8;
    copy_b128_to_lds(Vb + (size_t)d * kN + m0 + off, &Vl[d * VP + off]);
  }
}

__global__ __launch_bounds__(256) void attn_kernel(
    const f16* __restrict__ Qh, const f16* __restrict__ Kh,
    const f16* __restrict__ Vt, float* __restrict__ out) {
  extern __shared__ f16 smem[];
  // No pointer aggregates from LDS symbols (lld can't relocate them):
  // scalar bases + index arithmetic only.
  f16* const Kt0 = smem;                        // 2 x kKtHalfs
  f16* const Vl0 = smem + 2 * kKtHalfs;         // 2 x kVlHalfs
  f16* const Pl  = smem + 2 * kKtHalfs + 2 * kVlHalfs;

  const int tid = threadIdx.x;
  const int lane = tid & 31, wave = tid >> 5;
  const int half = lane >> 4, l16 = lane & 15;
  const int kb = half * 8;
  const int b = blockIdx.y, z = blockIdx.z;

  const size_t qsb = (size_t)(z * kB + b);        // query/value stream
  const size_t ksb = (size_t)((1 - z) * kB + b);  // key stream (cross)
  const f16* Qb = Qh + qsb * (size_t)kN * kC;
  const f16* Kb = Kh + ksb * (size_t)kN * kC;
  const f16* Vb = Vt + qsb * (size_t)kC * kN;
  float* ob = out + qsb * (size_t)kC * kN;

  const int n0 = blockIdx.x * 128 + wave * 16;

  // Load Q fragments for this wave's 16 rows (reused for all key tiles).
  v16h qf[8];
  {
    const f16* qr = Qb + (size_t)(n0 + l16) * kC;
#pragma unroll
    for (int ks = 0; ks < 8; ++ks) {
      int cb = ks * 32 + kb;
      qf[ks] = make_frag(qr + cb, qr + cb + 16);
    }
  }

  float mrow[8], lrow[8], scale[8];
  v8f oacc[16];
#pragma unroll
  for (int r = 0; r < 8; ++r) { mrow[r] = -__builtin_inff(); lrow[r] = 0.f; }
#pragma unroll
  for (int dt = 0; dt < 16; ++dt)
#pragma unroll
    for (int r = 0; r < 8; ++r) oacc[dt][r] = 0.f;

  f16* pw = Pl + wave * (16 * PP);

  // Prologue: start async fill of tile 0 into buffer 0.
  stage_tiles(Kb, Vb, 0, Kt0, Vl0, tid);

  constexpr int kTiles = kN / 32;
  for (int it = 0; it < kTiles; ++it) {
    const int m0 = it * 32;
    const int cur = it & 1, nxt = cur ^ 1;
    const f16* Kt = Kt0 + cur * kKtHalfs;
    const f16* Vl = Vl0 + cur * kVlHalfs;

    // Current buffer's async fill (issued last iteration by this wave) done,
    // then barrier so every wave's portion is visible.
    wait_async_copies();
    __syncthreads();

    // Kick off next tile's fill into the other buffer (reads of that buffer
    // from the previous iteration are complete past the barrier above).
    if (it + 1 < kTiles) {
      stage_tiles(Kb, Vb, m0 + 32, Kt0 + nxt * kKtHalfs, Vl0 + nxt * kVlHalfs,
                  tid);
      if (m0 + 64 < kN)
        __builtin_prefetch(Kb + (size_t)(m0 + 64) * kC, 0, 0);
    }

    // S = Q K^T : 16 rows x 32 keys, two 16-col WMMA tiles, K=c loop.
    v8f sa0 = {}, sa1 = {};
#pragma unroll
    for (int ks = 0; ks < 8; ++ks) {
      int cb = ks * 32 + kb;
      v16h b0 = make_frag(&Kt[l16 * KP + cb], &Kt[l16 * KP + cb + 16]);
      sa0 = __builtin_amdgcn_wmma_f32_16x16x32_f16(
          false, qf[ks], false, b0, (short)0, sa0, false, false);
      v16h b1 = make_frag(&Kt[(16 + l16) * KP + cb],
                          &Kt[(16 + l16) * KP + cb + 16]);
      sa1 = __builtin_amdgcn_wmma_f32_16x16x32_f16(
          false, qf[ks], false, b1, (short)0, sa1, false, false);
    }

    // Online softmax. Row = r + 8*half; this lane holds cols l16 and 16+l16.
#pragma unroll
    for (int r = 0; r < 8; ++r) {
      float v = fmaxf(sa0[r], sa1[r]);
#pragma unroll
      for (int off = 8; off >= 1; off >>= 1)
        v = fmaxf(v, __shfl_xor(v, off, 32));  // reduce within 16-lane half
      float mn = fmaxf(mrow[r], v);
      scale[r] = __expf(mrow[r] - mn);
      mrow[r] = mn;
      float p0 = __expf(sa0[r] - mn);
      float p1 = __expf(sa1[r] - mn);
      float rs = p0 + p1;
#pragma unroll
      for (int off = 8; off >= 1; off >>= 1) rs += __shfl_xor(rs, off, 32);
      lrow[r] = lrow[r] * scale[r] + rs;
      int M = r + half * 8;
      pw[M * PP + l16] = (f16)p0;  // D-layout -> row-major scratch
      pw[M * PP + 16 + l16] = (f16)p1;
    }
#pragma unroll
    for (int dt = 0; dt < 16; ++dt)
#pragma unroll
      for (int r = 0; r < 8; ++r) oacc[dt][r] *= scale[r];

    // Make P's LDS stores visible to this wave's A-fragment reads.
    asm volatile("s_wait_dscnt 0" ::: "memory");
    v16h pf = make_frag(&pw[l16 * PP + kb], &pw[l16 * PP + kb + 16]);

    // O += P V : K = 32 keys in one WMMA, 16 d-tiles.
#pragma unroll
    for (int dt = 0; dt < 16; ++dt) {
      v16h vf = make_frag(&Vl[(dt * 16 + l16) * VP + kb],
                          &Vl[(dt * 16 + l16) * VP + kb + 16]);
      oacc[dt] = __builtin_amdgcn_wmma_f32_16x16x32_f16(
          false, pf, false, vf, (short)0, oacc[dt], false, false);
    }
    // No trailing barrier: next iteration's top barrier orders buffer reuse.
  }

  // Epilogue: O /= rowsum; out[z][b][d][n], rows r are consecutive n.
#pragma unroll
  for (int dt = 0; dt < 16; ++dt) {
    int d = dt * 16 + l16;
    float4 v0, v1;
    v0.x = oacc[dt][0] / lrow[0]; v0.y = oacc[dt][1] / lrow[1];
    v0.z = oacc[dt][2] / lrow[2]; v0.w = oacc[dt][3] / lrow[3];
    v1.x = oacc[dt][4] / lrow[4]; v1.y = oacc[dt][5] / lrow[5];
    v1.z = oacc[dt][6] / lrow[6]; v1.w = oacc[dt][7] / lrow[7];
    float* p = ob + (size_t)d * kN + n0 + half * 8;
    *(float4*)p = v0;
    *(float4*)(p + 4) = v1;
  }
}

// ---------------------------------------------------------------------------
extern "C" void kernel_launch(void* const* d_in, const int* in_sizes, int n_in,
                              void* d_out, int out_size, void* d_ws,
                              size_t ws_size, hipStream_t stream) {
  const float* x1 = (const float*)d_in[0];
  const float* x2 = (const float*)d_in[1];
  const float* Wq = (const float*)d_in[2];
  const float* bq = (const float*)d_in[3];
  const float* Wk = (const float*)d_in[4];
  const float* bk = (const float*)d_in[5];
  const float* Wv = (const float*)d_in[6];
  const float* bv = (const float*)d_in[7];
  float* out = (float*)d_out;

  // Workspace layout (f16): Wh[3*256*256] | Qh | Kh | Vt (each 2*B*N*C).
  char* ws = (char*)d_ws;
  f16* Wh = (f16*)ws; ws += (size_t)3 * kC * kC * sizeof(f16);
  f16* Qh = (f16*)ws; ws += (size_t)2 * kB * kN * kC * sizeof(f16);
  f16* Kh = (f16*)ws; ws += (size_t)2 * kB * kN * kC * sizeof(f16);
  f16* Vt = (f16*)ws;

  wcvt_kernel<<<dim3(768), dim3(256), 0, stream>>>(Wq, Wk, Wv, Wh);
  proj_kernel<<<dim3(kN / 64, kB, 2), dim3(128), 0, stream>>>(
      x1, x2, Wh, bq, bk, bv, Qh, Kh, Vt);
  attn_kernel<<<dim3(kN / 128, kB, 2), dim3(256), kSmemBytes, stream>>>(
      Qh, Kh, Vt, out);
}
